// Self_attention_residuel_77257871720942
// MI455X (gfx1250) — matrix operand order repaired and checked by reading, not compiled, not run on hardware
//
#include <hip/hip_runtime.h>
#include <hip/hip_bf16.h>
#include <stdint.h>

// ---------------------------------------------------------------------------
// Self-attention + residual + LN for MI455X (gfx1250, wave32, WMMA).
// bf16 operands, fp32 WMMA accumulation (v_wmma_f32_16x16x32_bf16).
// GEMM mainloop: double-buffered LDS fed by GLOBAL_LOAD_ASYNC_TO_LDS_B128
// (ASYNCcnt pipeline), 64x64 WMMA tile per wave.
// Requires ws_size >= 136 MB + 64 KB.
// ---------------------------------------------------------------------------

#define B_BATCH 4
#define T_SEQ   2048
#define D_DIM   1024

typedef __attribute__((ext_vector_type(16))) __bf16 v16bf;
typedef __attribute__((ext_vector_type(8)))  float  v8f;

union Frag { v16bf v; uint4 q[2]; };

__device__ __forceinline__ uint16_t f2bf(float f) {
  uint32_t u = __builtin_bit_cast(uint32_t, f);
  return (uint16_t)((u + 0x7FFFu + ((u >> 16) & 1u)) >> 16);  // RNE
}

// Async copy 16B per lane: memory -> LDS, tracked by ASYNCcnt (no VGPR round
// trip).  the LDS address comes from the vdst VGPR; generic pointers to __shared__ carry
// the LDS byte offset in their low 32 bits on gfx1250.
__device__ __forceinline__ void async_b128(uint16_t* lds, const uint16_t* g) {
  uint32_t loff = (uint32_t)(size_t)lds;
  asm volatile("global_load_async_to_lds_b128 %0, %1, off"
               :: "v"(loff), "v"(g) : "memory");
}

// --------------------------- weight cast f32 -> bf16 -----------------------
__global__ __launch_bounds__(256) void cast_kernel(const float* __restrict__ src,
                                                   uint16_t* __restrict__ dst, int n4) {
  int i = blockIdx.x * 256 + threadIdx.x;
  if (i < n4) {
    float4 f = ((const float4*)src)[i];
    ushort4 o;
    o.x = f2bf(f.x); o.y = f2bf(f.y); o.z = f2bf(f.z); o.w = f2bf(f.w);
    ((ushort4*)dst)[i] = o;
  }
}

// --------------------------- LayerNorm -> bf16 -----------------------------
__global__ __launch_bounds__(256) void ln_kernel(const float* __restrict__ x,
                                                 const float* __restrict__ gamma,
                                                 const float* __restrict__ beta,
                                                 uint16_t* __restrict__ xn) {
  size_t row = blockIdx.x;                         // b*T + t
  int tid = threadIdx.x;                           // 256 threads, 4 elems each
  float4 xv = ((const float4*)(x + row * D_DIM))[tid];
  __shared__ float r1[256], r2[256];
  r1[tid] = xv.x + xv.y + xv.z + xv.w;
  r2[tid] = xv.x * xv.x + xv.y * xv.y + xv.z * xv.z + xv.w * xv.w;
  __syncthreads();
  for (int off = 128; off > 0; off >>= 1) {
    if (tid < off) { r1[tid] += r1[tid + off]; r2[tid] += r2[tid + off]; }
    __syncthreads();
  }
  float mu  = r1[0] * (1.0f / D_DIM);
  float var = r2[0] * (1.0f / D_DIM) - mu * mu;
  float inv = rsqrtf(var + 1e-5f);
  float4 g  = ((const float4*)gamma)[tid];
  float4 bt = ((const float4*)beta)[tid];
  ushort4 o;
  o.x = f2bf((xv.x - mu) * inv * g.x + bt.x);
  o.y = f2bf((xv.y - mu) * inv * g.y + bt.y);
  o.z = f2bf((xv.z - mu) * inv * g.z + bt.z);
  o.w = f2bf((xv.w - mu) * inv * g.w + bt.w);
  ((ushort4*)(xn + row * D_DIM))[tid] = o;
}

// ------------------ generic bf16 GEMM: C = A[M,K] @ Bt[N,K]^T --------------
// 256x128 block tile, BK=32, 8 waves in a 4x2 grid; each wave owns a 64x64
// sub-tile = 4x4 WMMA 16x16 tiles.  Double-buffered LDS via async-to-LDS.
// EPI: 0=bf16 store, 1=f32*scale, 2=bf16(acc+residual), 3=f32 relu(acc+bias).
#define BM 256
#define BN 128
#define BK 32
#define AU4 (BM * BK / 8)                 // 1024 uint4 per A stage
#define NU4 ((BM * BK + BN * BK) / 8)     // 1536 uint4 per stage total

template <int EPI>
__global__ __launch_bounds__(256) void gemm_bf16(
    const uint16_t* __restrict__ A, long long strideA, int lda,
    const uint16_t* __restrict__ Bt, long long strideB, int ldb,
    void* __restrict__ Cv, long long strideC, int ldc,
    int K, float scale,
    const float* __restrict__ res, long long strideR,
    const float* __restrict__ bias) {
  __shared__ uint16_t lA[2][BM * BK];
  __shared__ uint16_t lB[2][BN * BK];

  const int b = blockIdx.z;
  A  += (long long)b * strideA;
  Bt += (long long)b * strideB;

  const int tid  = threadIdx.x;
  const int wave = tid >> 5, lane = tid & 31;
  const int wm = wave >> 1, wn = wave & 1;       // 4 x 2 wave grid, 64x64 each
  const int hl = lane >> 4, rl = lane & 15;
  const int m0 = blockIdx.x * BM, n0 = blockIdx.y * BN;

  v8f acc[4][4];
  v8f zero = {};
  for (int mi = 0; mi < 4; mi++)
    for (int ni = 0; ni < 4; ni++) acc[mi][ni] = zero;

  // Issue one K-stage of async memory->LDS copies (6 per thread -> 6/wave).
  auto issue_stage = [&](int k0, int buf) {
    for (int u = tid; u < NU4; u += 256) {
      if (u < AU4) {
        int row = u >> 2, ce = (u & 3) * 8;
        async_b128(&lA[buf][row * BK + ce],
                   A + (size_t)(m0 + row) * lda + k0 + ce);
      } else {
        int w = u - AU4;
        int row = w >> 2, ce = (w & 3) * 8;
        async_b128(&lB[buf][row * BK + ce],
                   Bt + (size_t)(n0 + row) * ldb + k0 + ce);
      }
    }
  };

  const int nk = K / BK;
  issue_stage(0, 0);

  for (int s = 0; s < nk; s++) {
    const int buf = s & 1;
    if (s + 2 < nk) {  // L2 prefetch two stages ahead (global_prefetch_b8)
      __builtin_prefetch(A  + (size_t)(m0 + (tid >> 2)) * lda + (s + 2) * BK, 0, 1);
      __builtin_prefetch(Bt + (size_t)(n0 + (tid >> 2)) * ldb + (s + 2) * BK, 0, 1);
    }
    if (s + 1 < nk) {
      issue_stage((s + 1) * BK, buf ^ 1);
      // ASYNCcnt decrements in order: <=6 outstanding => stage s fully in LDS.
      asm volatile("s_wait_asynccnt 6" ::: "memory");
    } else {
      asm volatile("s_wait_asynccnt 0" ::: "memory");
    }
    __syncthreads();  // publish stage s LDS data to all waves

    // B fragments: lanes 0-15 hold K 0-15, lanes 16-31 hold K 16-31 (row = N)
    Frag bfr[4];
    for (int ni = 0; ni < 4; ni++) {
      const uint16_t* pb = &lB[buf][(wn * 64 + ni * 16 + rl) * BK + hl * 16];
      bfr[ni].q[0] = *(const uint4*)(pb);
      bfr[ni].q[1] = *(const uint4*)(pb + 8);
    }
    // A fragments: lanes 0-15 hold K 0-15, lanes 16-31 hold K 16-31 (row = M)
    for (int mi = 0; mi < 4; mi++) {
      Frag afr;
      const uint16_t* pa = &lA[buf][(wm * 64 + mi * 16 + rl) * BK + hl * 16];
      afr.q[0] = *(const uint4*)(pa);
      afr.q[1] = *(const uint4*)(pa + 8);
      for (int ni = 0; ni < 4; ni++) {
        acc[mi][ni] = __builtin_amdgcn_wmma_f32_16x16x32_bf16(
            false, afr.v, false, bfr[ni].v, (short)0, acc[mi][ni], false, false);
      }
    }
    __syncthreads();  // all waves done reading buf before it is refilled
  }

  // Epilogue.  C/D layout: VGPR r -> M = r + 8*hl, N = lane&15.
  const int ncol = rl;
  const int moff = hl * 8;
  for (int mi = 0; mi < 4; mi++) {
    for (int ni = 0; ni < 4; ni++) {
      int row = m0 + wm * 64 + mi * 16 + moff;
      int col = n0 + wn * 64 + ni * 16 + ncol;
      for (int r = 0; r < 8; r++) {
        float va = acc[mi][ni][r];
        size_t off = (size_t)(row + r) * ldc + col;
        if (EPI == 0) {
          ((uint16_t*)Cv)[(size_t)b * strideC + off] = f2bf(va);
        } else if (EPI == 1) {
          ((float*)Cv)[(size_t)b * strideC + off] = va * scale;
        } else if (EPI == 2) {
          float xr = res[(size_t)b * strideR + off];
          ((uint16_t*)Cv)[(size_t)b * strideC + off] = f2bf(va + xr);
        } else {
          float o = va + bias[col];
          ((float*)Cv)[(size_t)b * strideC + off] = fmaxf(o, 0.0f);
        }
      }
    }
  }
}

// ---------- column softmax stats over i (contiguous axis of scoresT) -------
__global__ __launch_bounds__(256) void colstats_kernel(const float* __restrict__ sT,
                                                       const int* __restrict__ l,
                                                       float* __restrict__ mOut,
                                                       float* __restrict__ sOut) {
  int j = blockIdx.x, b = blockIdx.y;
  const float* row = sT + ((size_t)b * T_SEQ + j) * T_SEQ;
  int L = l[b];
  int tid = threadIdx.x;
  __shared__ float red[256];
  float mx = -3.0e38f;
  for (int i = tid; i < L; i += 256) mx = fmaxf(mx, row[i]);
  red[tid] = mx; __syncthreads();
  for (int off = 128; off > 0; off >>= 1) {
    if (tid < off) red[tid] = fmaxf(red[tid], red[tid + off]);
    __syncthreads();
  }
  float m = red[0]; __syncthreads();
  float sm = 0.0f;
  for (int i = tid; i < L; i += 256) sm += __expf(row[i] - m);
  red[tid] = sm; __syncthreads();
  for (int off = 128; off > 0; off >>= 1) {
    if (tid < off) red[tid] += red[tid + off];
    __syncthreads();
  }
  if (tid == 0) { mOut[b * T_SEQ + j] = m; sOut[b * T_SEQ + j] = red[0]; }
}

// -------- transpose + normalize: attn[b,i,j] = mask(i) * exp(...)/sum ------
__global__ __launch_bounds__(256) void tnorm_kernel(const float* __restrict__ sT,
                                                    const float* __restrict__ mBuf,
                                                    const float* __restrict__ sBuf,
                                                    const int* __restrict__ l,
                                                    uint16_t* __restrict__ attn) {
  __shared__ float tile[32][33];
  int b = blockIdx.z;
  int i0 = blockIdx.x * 32, j0 = blockIdx.y * 32;
  int tx = threadIdx.x, ty = threadIdx.y;      // (32, 8)
  const float* base = sT + (size_t)b * T_SEQ * T_SEQ;
  for (int kk = 0; kk < 4; kk++)
    tile[ty + 8 * kk][tx] = base[(size_t)(j0 + ty + 8 * kk) * T_SEQ + (i0 + tx)];
  __syncthreads();
  int L = l[b];
  int j = j0 + tx;
  float mj  = mBuf[b * T_SEQ + j];
  float inv = 1.0f / sBuf[b * T_SEQ + j];
  uint16_t* ab = attn + (size_t)b * T_SEQ * T_SEQ;
  for (int kk = 0; kk < 4; kk++) {
    int i = i0 + ty + 8 * kk;
    float vv = tile[tx][ty + 8 * kk];
    float a = (i < L) ? __expf(vv - mj) * inv : 0.0f;
    ab[(size_t)i * T_SEQ + j] = f2bf(a);
  }
}

// ------------------------- V transpose (bf16) ------------------------------
__global__ __launch_bounds__(256) void vtrans_kernel(const uint16_t* __restrict__ v,
                                                     uint16_t* __restrict__ vT) {
  __shared__ uint16_t tile[32][33];
  int b = blockIdx.z;
  int d0 = blockIdx.x * 32, j0 = blockIdx.y * 32;
  int tx = threadIdx.x, ty = threadIdx.y;
  const uint16_t* vb = v + (size_t)b * T_SEQ * D_DIM;
  uint16_t* vtb = vT + (size_t)b * D_DIM * T_SEQ;
  for (int kk = 0; kk < 4; kk++)
    tile[ty + 8 * kk][tx] = vb[(size_t)(j0 + ty + 8 * kk) * D_DIM + (d0 + tx)];
  __syncthreads();
  for (int kk = 0; kk < 4; kk++)
    vtb[(size_t)(d0 + ty + 8 * kk) * T_SEQ + (j0 + tx)] = tile[tx][ty + 8 * kk];
}

// ---------------------------------------------------------------------------
extern "C" void kernel_launch(void* const* d_in, const int* in_sizes, int n_in,
                              void* d_out, int out_size, void* d_ws, size_t ws_size,
                              hipStream_t stream) {
  const float* x     = (const float*)d_in[0];
  const int*   l     = (const int*)d_in[1];
  const float* Wq    = (const float*)d_in[2];
  const float* Wk    = (const float*)d_in[3];
  const float* Wv    = (const float*)d_in[4];
  const float* Wo    = (const float*)d_in[5];
  const float* bo    = (const float*)d_in[6];
  const float* gamma = (const float*)d_in[7];
  const float* beta  = (const float*)d_in[8];
  float* out = (float*)d_out;

  char* ws = (char*)d_ws;
  const size_t MB = 1ull << 20;
  uint16_t* xn   = (uint16_t*)(ws + 0);        // 16MB, dead after QKV
  uint16_t* qb   = (uint16_t*)(ws + 16 * MB);  // 16MB, dead after scores
  uint16_t* kb   = (uint16_t*)(ws + 32 * MB);  // 16MB, dead after scores
  uint16_t* vb   = (uint16_t*)(ws + 48 * MB);  // 16MB
  float*    sT   = (float*)   (ws + 64 * MB);  // 64MB, dead after tnorm
  uint16_t* attn = (uint16_t*)(ws + 0);        // 32MB, overlays xn+q
  uint16_t* vT   = (uint16_t*)(ws + 32 * MB);  // 16MB, overlays k
  uint16_t* yb   = (uint16_t*)(ws + 64 * MB);  // 16MB, overlays sT
  uint16_t* wqb  = (uint16_t*)(ws + 128 * MB);
  uint16_t* wkb  = (uint16_t*)(ws + 130 * MB);
  uint16_t* wvb  = (uint16_t*)(ws + 132 * MB);
  uint16_t* wob  = (uint16_t*)(ws + 134 * MB);
  float*    mBuf = (float*)   (ws + 136 * MB);
  float*    sBuf = (float*)   (ws + 136 * MB + 32 * 1024);

  const long long TD = (long long)T_SEQ * D_DIM;
  const long long TT = (long long)T_SEQ * T_SEQ;
  const long long DT = (long long)D_DIM * T_SEQ;

  // weights -> bf16
  int n4 = D_DIM * D_DIM / 4;
  cast_kernel<<<n4 / 256, 256, 0, stream>>>(Wq, wqb, n4);
  cast_kernel<<<n4 / 256, 256, 0, stream>>>(Wk, wkb, n4);
  cast_kernel<<<n4 / 256, 256, 0, stream>>>(Wv, wvb, n4);
  cast_kernel<<<n4 / 256, 256, 0, stream>>>(Wo, wob, n4);

  // LayerNorm
  ln_kernel<<<B_BATCH * T_SEQ, 256, 0, stream>>>(x, gamma, beta, xn);

  // Q,K,V = xn @ W^T   (M=8192, N=1024, K=1024)
  dim3 gq(8192 / BM, 1024 / BN, 1);
  gemm_bf16<0><<<gq, 256, 0, stream>>>(xn, 0, D_DIM, wqb, 0, D_DIM, qb, 0, D_DIM,
                                       D_DIM, 1.0f, nullptr, 0, nullptr);
  gemm_bf16<0><<<gq, 256, 0, stream>>>(xn, 0, D_DIM, wkb, 0, D_DIM, kb, 0, D_DIM,
                                       D_DIM, 1.0f, nullptr, 0, nullptr);
  gemm_bf16<0><<<gq, 256, 0, stream>>>(xn, 0, D_DIM, wvb, 0, D_DIM, vb, 0, D_DIM,
                                       D_DIM, 1.0f, nullptr, 0, nullptr);

  // scoresT[b,j,i] = (K @ Q^T) / 32   (per batch: M=N=2048, K=1024)
  dim3 gs(2048 / BM, 2048 / BN, B_BATCH);
  gemm_bf16<1><<<gs, 256, 0, stream>>>(kb, TD, D_DIM, qb, TD, D_DIM,
                                       sT, TT, T_SEQ, D_DIM, 0.03125f, nullptr, 0, nullptr);

  // softmax over i (dim 1) per column j, masked to i < l[b]
  dim3 gst(T_SEQ, B_BATCH);
  colstats_kernel<<<gst, 256, 0, stream>>>(sT, l, mBuf, sBuf);

  dim3 btn(32, 8);
  dim3 gtn(T_SEQ / 32, T_SEQ / 32, B_BATCH);
  tnorm_kernel<<<gtn, btn, 0, stream>>>(sT, mBuf, sBuf, l, attn);

  // V^T for the attn@V GEMM (B-operand must be [N,K] K-contiguous)
  dim3 gvt(D_DIM / 32, T_SEQ / 32, B_BATCH);
  vtrans_kernel<<<gvt, btn, 0, stream>>>(vb, vT);

  // y = x + attn @ V   (per batch: M=2048, N=1024, K=2048)
  dim3 gav(2048 / BM, 1024 / BN, B_BATCH);
  gemm_bf16<2><<<gav, 256, 0, stream>>>(attn, TT, T_SEQ, vT, DT, T_SEQ,
                                        yb, TD, D_DIM, T_SEQ, 1.0f, x, TD, nullptr);

  // out = relu(y @ Wo^T + bo)   (M=8192, N=1024, K=1024)
  dim3 go(8192 / BM, 1024 / BN, 1);
  gemm_bf16<3><<<go, 256, 0, stream>>>(yb, 0, D_DIM, wob, 0, D_DIM, out, 0, D_DIM,
                                       D_DIM, 1.0f, nullptr, 0, bo);
}
